// GCN_75763223102093
// MI455X (gfx1250) — matrix operand order repaired and checked by reading, not compiled, not run on hardware
//
#include <hip/hip_runtime.h>
#include <hip/hip_bf16.h>
#include <math.h>

#define N_NODES   100000
#define N_EDGES   1600000
#define N_FEAT    512
#define N_HID     256
#define N_CLASS   40
#define N_CLASS_P 48   // padded to 3 WMMA n-tiles

typedef _Float16 half8  __attribute__((ext_vector_type(8)));
typedef _Float16 half16 __attribute__((ext_vector_type(16)));
typedef float    f32x8  __attribute__((ext_vector_type(8)));

union H16 { _Float16 h[16]; half16 v; };

__device__ __forceinline__ void split_f32(float f, _Float16& hi, _Float16& lo) {
    _Float16 h = (_Float16)f;
    hi = h;
    lo = (_Float16)(f - (float)h);
}

__device__ __forceinline__ void atomic_add_f32(float* p, float v) {
    unsafeAtomicAdd(p, v);   // expect global_atomic_add_f32 on gfx1250
}

// ---------------------------------------------------------------------------
// Prep: transpose + split weights into f16 hi/lo, [N,K] row-major (K contig).
// W1: [512,256] fp32 -> W1T*: [256,512] f16.  W2: [256,40] -> [48,256], pad 0.
// ---------------------------------------------------------------------------
__global__ void prep_weights(const float* __restrict__ W1, const float* __restrict__ W2,
                             _Float16* __restrict__ W1Thi, _Float16* __restrict__ W1Tlo,
                             _Float16* __restrict__ W2Thi, _Float16* __restrict__ W2Tlo) {
    int idx = blockIdx.x * blockDim.x + threadIdx.x;
    const int n1 = N_HID * N_FEAT;      // 131072
    const int n2 = N_CLASS_P * N_HID;   // 12288
    if (idx < n1) {
        int n = idx / N_FEAT, k = idx % N_FEAT;
        _Float16 hi, lo;
        split_f32(W1[k * N_HID + n], hi, lo);
        W1Thi[idx] = hi; W1Tlo[idx] = lo;
    } else if (idx < n1 + n2) {
        int j = idx - n1;
        int n = j / N_HID, k = j % N_HID;
        float v = (n < N_CLASS) ? W2[k * N_CLASS + n] : 0.0f;
        _Float16 hi, lo;
        split_f32(v, hi, lo);
        W2Thi[j] = hi; W2Tlo[j] = lo;
    }
}

__global__ void zero_bufs(float* __restrict__ H2, float* __restrict__ out) {
    long i = (long)blockIdx.x * blockDim.x + threadIdx.x;
    long stride = (long)gridDim.x * blockDim.x;
    const long nh = (long)N_NODES * N_HID / 4;    // float4 counts
    const long no = (long)N_NODES * N_CLASS / 4;
    float4 z = make_float4(0.f, 0.f, 0.f, 0.f);
    for (long t = i; t < nh; t += stride) ((float4*)H2)[t] = z;
    for (long t = i; t < no; t += stride) ((float4*)out)[t] = z;
}

// ---------------------------------------------------------------------------
// Split-precision f16 WMMA GEMM:  D[M,NOUT] = act(A)[M,K] * B[K,NOUT] + bias
// One wave per 16x16 output tile; 8 waves (tiles) per block.
// BT* are f16 [NOUTP, K] (pre-transposed hi/lo).
// ---------------------------------------------------------------------------
template<int K, int NOUT, int NOUTP, bool RELU>
__global__ void __launch_bounds__(256)
gemm_split_f16(const float* __restrict__ A,
               const _Float16* __restrict__ BThi, const _Float16* __restrict__ BTlo,
               const float* __restrict__ bias, float* __restrict__ D) {
    const int lane = threadIdx.x & 31;
    const int wid  = threadIdx.x >> 5;
    const int NTN  = NOUTP / 16;
    const int NT   = (N_NODES / 16) * NTN;
    int tile = blockIdx.x * 8 + wid;
    if (tile >= NT) return;
    int mt = tile / NTN, nt = tile % NTN;
    int m0 = mt * 16, n0 = nt * 16;
    int hlf = lane >> 4;          // 0 or 1 (lane group)
    int ln  = lane & 15;          // A row / B col within tile
    int koffA = hlf * 8;
    int koffB = hlf * 8;

    const float*    arow = A    + (long)(m0 + ln) * K;
    const _Float16* bh   = BThi + (long)(n0 + ln) * K;
    const _Float16* bl   = BTlo + (long)(n0 + ln) * K;

    int col = n0 + ln;
    float bv = (col < NOUT) ? bias[col] : 0.0f;
    f32x8 acc;
#pragma unroll
    for (int i = 0; i < 8; ++i) acc[i] = bv;

#pragma unroll 2
    for (int kb = 0; kb < K; kb += 32) {
        // --- A fragment: fp32 load, on-the-fly hi/lo split ---
        const float4* ap = (const float4*)(arow + kb + koffA);
        float4 a0 = ap[0], a1 = ap[1], a2 = ap[4], a3 = ap[5];
        float av[16] = { a0.x, a0.y, a0.z, a0.w,  a1.x, a1.y, a1.z, a1.w,
                         a2.x, a2.y, a2.z, a2.w,  a3.x, a3.y, a3.z, a3.w };
        H16 ahi, alo;
#pragma unroll
        for (int e = 0; e < 16; ++e) {
            float f = RELU ? fmaxf(av[e], 0.0f) : av[e];
            split_f32(f, ahi.h[e], alo.h[e]);
        }
        // --- B fragment: two runs of 8 f16 per lane at k=8h and k=16+8h (pre-transposed) ---
        half8 b0 = *(const half8*)(bh + kb + koffB);
        half8 b1 = *(const half8*)(bh + kb + koffB + 16);
        half8 c0 = *(const half8*)(bl + kb + koffB);
        half8 c1 = *(const half8*)(bl + kb + koffB + 16);
        half16 bhi = __builtin_shufflevector(b0, b1, 0,1,2,3,4,5,6,7,8,9,10,11,12,13,14,15);
        half16 blo = __builtin_shufflevector(c0, c1, 0,1,2,3,4,5,6,7,8,9,10,11,12,13,14,15);

        // D += Ahi*Bhi + Ahi*Blo + Alo*Bhi   (fp32-class accuracy)
        acc = __builtin_amdgcn_wmma_f32_16x16x32_f16(false, ahi.v, false, bhi, (short)0, acc, false, false);
        acc = __builtin_amdgcn_wmma_f32_16x16x32_f16(false, ahi.v, false, blo, (short)0, acc, false, false);
        acc = __builtin_amdgcn_wmma_f32_16x16x32_f16(false, alo.v, false, bhi, (short)0, acc, false, false);
    }

    if (col < NOUT) {
        float* orow = D + (long)m0 * NOUT + col;
#pragma unroll
        for (int i = 0; i < 8; ++i)
            orow[(long)(i + hlf * 8) * NOUT] = acc[i];
    }
}

// ---------------------------------------------------------------------------
// SPMM scatter, one wave per edge.  out[row[e]] += w[e] * src[col[e]]
// ---------------------------------------------------------------------------
__global__ void __launch_bounds__(256)
scatter_hid(const int* __restrict__ erow, const int* __restrict__ ecol,
            const float* __restrict__ ew, const float* __restrict__ src,
            float* __restrict__ dst) {
    int lane = threadIdx.x & 31;
    int e = blockIdx.x * 8 + (threadIdx.x >> 5);
    if (e >= N_EDGES) return;
    int r = erow[e], c = ecol[e];
    float w = ew[e];
    const float4* s = (const float4*)(src + (long)c * N_HID) + lane * 2;
    float4 u = s[0], v = s[1];
    float* d = dst + (long)r * N_HID + lane * 8;
    atomic_add_f32(d + 0, w * u.x);
    atomic_add_f32(d + 1, w * u.y);
    atomic_add_f32(d + 2, w * u.z);
    atomic_add_f32(d + 3, w * u.w);
    atomic_add_f32(d + 4, w * v.x);
    atomic_add_f32(d + 5, w * v.y);
    atomic_add_f32(d + 6, w * v.z);
    atomic_add_f32(d + 7, w * v.w);
}

__global__ void __launch_bounds__(256)
scatter_cls(const int* __restrict__ erow, const int* __restrict__ ecol,
            const float* __restrict__ ew, const float* __restrict__ src,
            float* __restrict__ dst) {
    int lane = threadIdx.x & 31;
    int e = blockIdx.x * 8 + (threadIdx.x >> 5);
    if (e >= N_EDGES) return;
    int r = erow[e], c = ecol[e];
    float w = ew[e];
    const float* s = src + (long)c * N_CLASS;
    float*       d = dst + (long)r * N_CLASS;
    atomic_add_f32(d + lane, w * s[lane]);          // lane 0..31 < 40
    if (lane < N_CLASS - 32) {
        int j = lane + 32;
        atomic_add_f32(d + j, w * s[j]);
    }
}

// ---------------------------------------------------------------------------
// In-place row-wise log_softmax over 40 classes, one wave per row.
// ---------------------------------------------------------------------------
__global__ void __launch_bounds__(256)
logsoftmax_rows(float* __restrict__ out) {
    int lane = threadIdx.x & 31;
    int r = blockIdx.x * 8 + (threadIdx.x >> 5);
    if (r >= N_NODES) return;
    float* rp = out + (long)r * N_CLASS;
    float x0 = rp[lane];
    bool has2 = lane < (N_CLASS - 32);
    float x1 = has2 ? rp[lane + 32] : -INFINITY;
    float m = fmaxf(x0, x1);
#pragma unroll
    for (int s = 16; s > 0; s >>= 1) m = fmaxf(m, __shfl_xor(m, s, 32));
    float sum = expf(x0 - m) + (has2 ? expf(x1 - m) : 0.0f);
#pragma unroll
    for (int s = 16; s > 0; s >>= 1) sum += __shfl_xor(sum, s, 32);
    float l = m + logf(sum);
    rp[lane] = x0 - l;
    if (has2) rp[lane + 32] = x1 - l;
}

// ---------------------------------------------------------------------------
extern "C" void kernel_launch(void* const* d_in, const int* in_sizes, int n_in,
                              void* d_out, int out_size, void* d_ws, size_t ws_size,
                              hipStream_t stream) {
    const float* X  = (const float*)d_in[0];
    const int*   ei = (const int*)  d_in[1];
    const float* ew = (const float*)d_in[2];
    const float* W1 = (const float*)d_in[3];
    const float* b1 = (const float*)d_in[4];
    const float* W2 = (const float*)d_in[5];
    const float* b2 = (const float*)d_in[6];
    const int* erow = ei;             // edge_index[0]
    const int* ecol = ei + N_EDGES;   // edge_index[1]
    float* out = (float*)d_out;

    char* ws = (char*)d_ws;
    const size_t oH1 = 0;                               // 100000*256*4 = 102,400,000
    const size_t oH2 = oH1 + 102400000ull;              // 102,400,000
    const size_t oH3 = oH2 + 102400000ull;              // 100000*40*4 = 16,000,000
    const size_t oW1h = oH3 + 16000000ull;              // 256*512*2 = 262,144
    const size_t oW1l = oW1h + 262144ull;
    const size_t oW2h = oW1l + 262144ull;               // 48*256*2 = 24,576
    const size_t oW2l = oW2h + 24576ull;
    float*    H1 = (float*)(ws + oH1);
    float*    H2 = (float*)(ws + oH2);
    float*    H3 = (float*)(ws + oH3);
    _Float16* W1Thi = (_Float16*)(ws + oW1h);
    _Float16* W1Tlo = (_Float16*)(ws + oW1l);
    _Float16* W2Thi = (_Float16*)(ws + oW2h);
    _Float16* W2Tlo = (_Float16*)(ws + oW2l);

    // 1) weight transpose + f16 hi/lo split; zero accumulators
    prep_weights<<<(N_HID * N_FEAT + N_CLASS_P * N_HID + 255) / 256, 256, 0, stream>>>(
        W1, W2, W1Thi, W1Tlo, W2Thi, W2Tlo);
    zero_bufs<<<2048, 256, 0, stream>>>(H2, out);

    // 2) H1 = X @ W1 + b1          (split-f16 WMMA, fp32 accumulate)
    gemm_split_f16<N_FEAT, N_HID, N_HID, false>
        <<<(N_NODES / 16) * (N_HID / 16) / 8, 256, 0, stream>>>(X, W1Thi, W1Tlo, b1, H1);

    // 3) H2[r] += w_e * H1[c]      (SPMM via float atomics)
    scatter_hid<<<N_EDGES / 8, 256, 0, stream>>>(erow, ecol, ew, H1, H2);

    // 4) H3 = relu(H2) @ W2 + b2   (relu folded into A-fragment load)
    gemm_split_f16<N_HID, N_CLASS, N_CLASS_P, true>
        <<<((N_NODES / 16) * (N_CLASS_P / 16) + 7) / 8, 256, 0, stream>>>(H2, W2Thi, W2Tlo, b2, H3);

    // 5) out[r] += w_e * H3[c]
    scatter_cls<<<N_EDGES / 8, 256, 0, stream>>>(erow, ecol, ew, H3, out);

    // 6) in-place log_softmax
    logsoftmax_rows<<<N_NODES / 8, 256, 0, stream>>>(out);
}
